// InformerEncoder_59760174956590
// MI455X (gfx1250) — matrix operand order repaired and checked by reading, not compiled, not run on hardware
//
#include <hip/hip_runtime.h>

typedef __bf16 bf16;
typedef __bf16 bf16x8  __attribute__((ext_vector_type(8)));
typedef __bf16 bf16x16 __attribute__((ext_vector_type(16)));
typedef float  f32x8   __attribute__((ext_vector_type(8)));
typedef int    v4i     __attribute__((vector_size(16)));   // matches builtin param

#define DM 512
#define NB 8

// Use the gfx1250 async global->LDS path when the toolchain exposes it.
#if defined(__has_builtin)
# if __has_builtin(__builtin_amdgcn_global_load_async_to_lds_b128) && \
     __has_builtin(__builtin_amdgcn_s_wait_asynccnt)
#  define USE_ASYNC_LDS 1
# endif
#endif
#ifndef USE_ASYNC_LDS
# define USE_ASYNC_LDS 0
#endif

#define AS1_V4I(p) ((__attribute__((address_space(1))) v4i*)(p))
#define AS3_V4I(p) ((__attribute__((address_space(3))) v4i*)(p))

// ---------------------------------------------------------------------------
// WMMA GEMM: C[M,N] = A[M,K] * B[N,K]^T + bias   (A,B bf16; C f32)
// A has row stride lda (elements); B is [N][K] contiguous (weight layout).
// grid = (M/128, N/128, Z) with per-Z strides for batched conv GEMM.
// 256 threads = 8 wave32; wave tile 32x64 = 2x4 WMMA 16x16x32 accums.
// Double-buffered LDS; tile t+1 staged (async-to-LDS or reg-staged) while
// tile t runs its 8 WMMAs.
// ---------------------------------------------------------------------------
__global__ __launch_bounds__(256) void gemm_wmma_bf16(
    const bf16* __restrict__ A, long aStride, int lda,
    const bf16* __restrict__ B,
    float* __restrict__ C, long cStride, int ldc,
    const float* __restrict__ bias, int K)
{
  __shared__ alignas(16) bf16 As[2][128][40];
  __shared__ alignas(16) bf16 Bs[2][128][40];

  const int tid  = threadIdx.x;
  const int lane = tid & 31;
  const int wave = tid >> 5;
  const int wm   = wave & 3;      // wave M position (0..3)
  const int wn   = wave >> 2;     // wave N position (0..1)
  const int kh   = lane >> 4;     // lane half (K split per ISA layout)
  const int lr   = lane & 15;

  const long tileM = (long)blockIdx.x * 128;
  const int  tileN = blockIdx.y * 128;
  const bf16* Ab = A + (long)blockIdx.z * aStride;
  float*      Cb = C + (long)blockIdx.z * cStride;

  // staging coordinates: each thread moves rows r0 and r0+64, one 16B chunk
  const int r0 = tid >> 2;
  const int r1 = r0 + 64;
  const int kc = (tid & 3) << 3;
  const bf16* gA0 = Ab + (tileM + r0) * (long)lda + kc;
  const bf16* gA1 = Ab + (tileM + r1) * (long)lda + kc;
  const bf16* gB0 = B + (long)(tileN + r0) * K + kc;
  const bf16* gB1 = B + (long)(tileN + r1) * K + kc;

  f32x8 acc[2][4] = {};

  // 8 WMMAs on one staged 128x32 tile pair
  auto compute_tile = [&](int b2) {
    // A fragment 16x32: lanes 0-15 K0-7/K16-23, lanes 16-31 K8-15/K24-31
    bf16x16 af[2];
#pragma unroll
    for (int mf = 0; mf < 2; ++mf) {
      int r = wm * 32 + mf * 16 + lr;
      bf16x8 lo = *(const bf16x8*)&As[b2][r][kh * 8];
      bf16x8 hi = *(const bf16x8*)&As[b2][r][kh * 8 + 16];
      af[mf] = __builtin_shufflevector(lo, hi, 0,1,2,3,4,5,6,7,8,9,10,11,12,13,14,15);
    }
    // B fragment 32x16: column = lane%16, lanes 0-15 K0-7/K16-23, lanes 16-31 K8-15/K24-31
    bf16x16 bfr[4];
#pragma unroll
    for (int nf = 0; nf < 4; ++nf) {
      int cidx = wn * 64 + nf * 16 + lr;
      bf16x8 lo = *(const bf16x8*)&Bs[b2][cidx][kh * 8];
      bf16x8 hi = *(const bf16x8*)&Bs[b2][cidx][kh * 8 + 16];
      bfr[nf] = __builtin_shufflevector(lo, hi, 0,1,2,3,4,5,6,7,8,9,10,11,12,13,14,15);
    }
#pragma unroll
    for (int mf = 0; mf < 2; ++mf)
#pragma unroll
      for (int nf = 0; nf < 4; ++nf)
        acc[mf][nf] = __builtin_amdgcn_wmma_f32_16x16x32_bf16(
            false, af[mf], false, bfr[nf], (short)0, acc[mf][nf], false, false);
  };

#if USE_ASYNC_LDS
  auto stage_async = [&](int b2, int k0) {
    __builtin_amdgcn_global_load_async_to_lds_b128(
        AS1_V4I(gA0 + k0), AS3_V4I(&As[b2][r0][kc]), 0, 0);
    __builtin_amdgcn_global_load_async_to_lds_b128(
        AS1_V4I(gA1 + k0), AS3_V4I(&As[b2][r1][kc]), 0, 0);
    __builtin_amdgcn_global_load_async_to_lds_b128(
        AS1_V4I(gB0 + k0), AS3_V4I(&Bs[b2][r0][kc]), 0, 0);
    __builtin_amdgcn_global_load_async_to_lds_b128(
        AS1_V4I(gB1 + k0), AS3_V4I(&Bs[b2][r1][kc]), 0, 0);
  };
  stage_async(0, 0);
  __builtin_amdgcn_s_wait_asynccnt(0);
  __syncthreads();
  int buf = 0;
  for (int k0 = 0; k0 < K; k0 += 32) {
    if (k0 + 32 < K) stage_async(buf ^ 1, k0 + 32);   // overlaps WMMAs below
    compute_tile(buf);
    __builtin_amdgcn_s_wait_asynccnt(0);
    __syncthreads();
    buf ^= 1;
  }
#else
  // register-staged pipeline: next tile's global loads issue before compute
  bf16x8 ra0 = *(const bf16x8*)gA0, ra1 = *(const bf16x8*)gA1;
  bf16x8 rb0 = *(const bf16x8*)gB0, rb1 = *(const bf16x8*)gB1;
  *(bf16x8*)&As[0][r0][kc] = ra0; *(bf16x8*)&As[0][r1][kc] = ra1;
  *(bf16x8*)&Bs[0][r0][kc] = rb0; *(bf16x8*)&Bs[0][r1][kc] = rb1;
  __syncthreads();
  int buf = 0;
  for (int k0 = 0; k0 < K; k0 += 32) {
    const bool more = (k0 + 32) < K;
    if (more) {
      ra0 = *(const bf16x8*)(gA0 + k0 + 32);
      ra1 = *(const bf16x8*)(gA1 + k0 + 32);
      rb0 = *(const bf16x8*)(gB0 + k0 + 32);
      rb1 = *(const bf16x8*)(gB1 + k0 + 32);
      __builtin_prefetch(gA0 + k0 + 64, 0, 0);
      __builtin_prefetch(gB0 + k0 + 64, 0, 0);
    }
    compute_tile(buf);          // loads above remain in flight during WMMAs
    if (more) {
      *(bf16x8*)&As[buf ^ 1][r0][kc] = ra0;
      *(bf16x8*)&As[buf ^ 1][r1][kc] = ra1;
      *(bf16x8*)&Bs[buf ^ 1][r0][kc] = rb0;
      *(bf16x8*)&Bs[buf ^ 1][r1][kc] = rb1;
    }
    __syncthreads();
    buf ^= 1;
  }
#endif

  // C/D layout: vgpr i -> M = (lane>>4)*8 + i, N = lane&15
#pragma unroll
  for (int mf = 0; mf < 2; ++mf) {
#pragma unroll
    for (int nf = 0; nf < 4; ++nf) {
      int  n  = tileN + wn * 64 + nf * 16 + lr;
      long mb = tileM + wm * 32 + mf * 16 + kh * 8;
      float bv = bias ? bias[n] : 0.0f;
#pragma unroll
      for (int i = 0; i < 8; ++i)
        Cb[(mb + i) * (long)ldc + n] = acc[mf][nf][i] + bv;
    }
  }
}

// ---------------------------------------------------------------------------
// Elementwise / small kernels
// ---------------------------------------------------------------------------
__global__ void f2bf_kernel(const float* __restrict__ s, bf16* __restrict__ d, long n) {
  long i = (long)blockIdx.x * blockDim.x + threadIdx.x;
  if (i < n) d[i] = (bf16)s[i];
}

__global__ void tok_embed_kernel(const float* __restrict__ xe,
                                 const float* __restrict__ tw,
                                 float* __restrict__ x, int L) {
  long i = (long)blockIdx.x * blockDim.x + threadIdx.x;
  long total = (long)NB * L * DM;
  if (i >= total) return;
  int  d  = (int)(i % DM);
  long bl = i / DM;
  int  l  = (int)(bl % L);
  int  b  = (int)(bl / L);
  int lm = (l - 1 + L) % L, lp = (l + 1) % L;
  const float* xr = xe + (long)b * L;
  float s = tw[d*3+0]*xr[lm] + tw[d*3+1]*xr[l] + tw[d*3+2]*xr[lp];
  int dh = d >> 1;
  float div = __expf((float)(2*dh) * (-9.210340371976184f / (float)DM));
  float ang = (float)l * div;
  s += (d & 1) ? __cosf(ang) : __sinf(ang);
  x[i] = s;
}

__device__ __forceinline__ int samp_idx(int l, int u, int layer, int L) {
  unsigned r = (unsigned)l * 2654435761u;
  r ^= (unsigned)u * 40503u + 0x9E3779B9u;
  r ^= (unsigned)layer * 2246822519u;
  r ^= r >> 13; r *= 0x85EBCA6Bu; r ^= r >> 16;
  return (int)(r & (unsigned)(L - 1));   // L is a power of two
}

// M-measure: max_u(q.k_samp) - mean_u(q.k_samp)
__global__ void meas_kernel(const float* __restrict__ q, const float* __restrict__ k,
                            float* __restrict__ Mm, int L, int U, int layer) {
  long i = (long)blockIdx.x * blockDim.x + threadIdx.x;
  long total = (long)64 * L;
  if (i >= total) return;
  int l = (int)(i % L); int bh = (int)(i / L);
  int h = bh & 7, b = bh >> 3;
  const float* qr = q + ((long)(b * L + l)) * DM + h * 64;
  float mx = -3.4e38f, sm = 0.f;
  for (int u = 0; u < U; ++u) {
    int kl = samp_idx(l, u, layer, L);
    const float* kr = k + ((long)(b * L + kl)) * DM + h * 64;
    float d0 = 0.f;
    for (int d = 0; d < 64; ++d) d0 += qr[d] * kr[d];
    mx = fmaxf(mx, d0); sm += d0;
  }
  Mm[i] = mx - sm / (float)U;
}

__global__ __launch_bounds__(256) void topk_kernel(const float* __restrict__ M,
                                                   int* __restrict__ top, int L, int U) {
  __shared__ float buf[2048];
  __shared__ float rv[256];
  __shared__ int   ri[256];
  int bh = blockIdx.x, tid = threadIdx.x;
  const float* Mr = M + (long)bh * L;
  for (int i = tid; i < L; i += 256) buf[i] = Mr[i];
  __syncthreads();
  for (int s = 0; s < U; ++s) {
    float best = -3.4e38f; int bi = 0;
    for (int i = tid; i < L; i += 256)
      if (buf[i] > best) { best = buf[i]; bi = i; }
    rv[tid] = best; ri[tid] = bi; __syncthreads();
    for (int o = 128; o > 0; o >>= 1) {
      if (tid < o && rv[tid + o] > rv[tid]) { rv[tid] = rv[tid + o]; ri[tid] = ri[tid + o]; }
      __syncthreads();
    }
    if (tid == 0) { top[(long)bh * U + s] = ri[0]; buf[ri[0]] = -3.4e38f; }
    __syncthreads();
  }
}

__global__ void vmean_kernel(const float* __restrict__ v, float* __restrict__ vm, int L) {
  int i = blockIdx.x * blockDim.x + threadIdx.x;
  if (i >= NB * DM) return;
  int dd = i % DM, b = i / DM;
  float s = 0.f;
  for (int l = 0; l < L; ++l) s += v[((long)(b * L + l)) * DM + dd];
  vm[i] = s / (float)L;
}

__global__ void ctxfill_kernel(const float* __restrict__ vm, float* __restrict__ o, int L) {
  long i = (long)blockIdx.x * blockDim.x + threadIdx.x;
  long total = (long)NB * L * DM;
  if (i >= total) return;
  int dd = (int)(i % DM);
  int b  = (int)((i / DM) / L);
  o[i] = vm[b * DM + dd];
}

__global__ __launch_bounds__(256) void attn_update_kernel(
    const float* __restrict__ q, const float* __restrict__ k, const float* __restrict__ v,
    const int* __restrict__ top, float* __restrict__ o, int L, int U) {
  __shared__ float sc[2048];
  __shared__ float qred[64];
  __shared__ float red[256];
  __shared__ float part[4][64];
  int tid = threadIdx.x;
  int u = blockIdx.x % U, bh = blockIdx.x / U;
  int h = bh & 7, b = bh >> 3;
  int tl = top[(long)bh * U + u];
  if (tid < 64) qred[tid] = q[((long)(b * L + tl)) * DM + h * 64 + tid];
  __syncthreads();
  float lmax = -3.4e38f;
  for (int kl = tid; kl < L; kl += 256) {
    const float* kr = k + ((long)(b * L + kl)) * DM + h * 64;
    float d0 = 0.f;
    for (int d = 0; d < 64; ++d) d0 += qred[d] * kr[d];
    d0 *= 0.125f;                      // 1/sqrt(64)
    sc[kl] = d0;
    lmax = fmaxf(lmax, d0);
  }
  red[tid] = lmax; __syncthreads();
  for (int off = 128; off > 0; off >>= 1) {
    if (tid < off) red[tid] = fmaxf(red[tid], red[tid + off]);
    __syncthreads();
  }
  float smax = red[0]; __syncthreads();
  float lsum = 0.f;
  for (int kl = tid; kl < L; kl += 256) { float e = __expf(sc[kl] - smax); sc[kl] = e; lsum += e; }
  red[tid] = lsum; __syncthreads();
  for (int off = 128; off > 0; off >>= 1) {
    if (tid < off) red[tid] += red[tid + off];
    __syncthreads();
  }
  float ssum = red[0]; __syncthreads();
  int d = tid & 63, sl = tid >> 6;
  float a = 0.f;
  for (int kl = sl; kl < L; kl += 4)
    a += sc[kl] * v[((long)(b * L + kl)) * DM + h * 64 + d];
  part[sl][d] = a; __syncthreads();
  if (tid < 64) {
    float s = (part[0][tid] + part[1][tid] + part[2][tid] + part[3][tid]) / ssum;
    o[((long)(b * L + tl)) * DM + h * 64 + tid] = s;
  }
}

// x_out = LayerNorm(x (+ resid)) * g + b   ; one block per row, dm = 512
__global__ __launch_bounds__(256) void resid_ln_kernel(
    const float* __restrict__ xin, const float* __restrict__ resid,
    float* __restrict__ xout, const float* __restrict__ g, const float* __restrict__ be) {
  __shared__ float red[256];
  long row = blockIdx.x;
  int tid = threadIdx.x;
  const float* xr = xin + row * DM;
  float t0 = xr[tid], t1 = xr[tid + 256];
  if (resid) { t0 += resid[row * DM + tid]; t1 += resid[row * DM + tid + 256]; }
  red[tid] = t0 + t1; __syncthreads();
  for (int o = 128; o > 0; o >>= 1) { if (tid < o) red[tid] += red[tid + o]; __syncthreads(); }
  float mu = red[0] / (float)DM; __syncthreads();
  float d0 = t0 - mu, d1 = t1 - mu;
  red[tid] = d0 * d0 + d1 * d1; __syncthreads();
  for (int o = 128; o > 0; o >>= 1) { if (tid < o) red[tid] += red[tid + o]; __syncthreads(); }
  float rstd = rsqrtf(red[0] / (float)DM + 1e-5f);
  xout[row * DM + tid]       = d0 * rstd * g[tid]       + be[tid];
  xout[row * DM + tid + 256] = d1 * rstd * g[tid + 256] + be[tid + 256];
}

__global__ void gelu_kernel(float* __restrict__ y, long n) {
  long i = (long)blockIdx.x * blockDim.x + threadIdx.x;
  if (i >= n) return;
  float x = y[i];
  y[i] = 0.5f * x * (1.0f + erff(x * 0.7071067811865476f));
}

// circular pad (1,1) along L and convert to bf16: xp is [B][L+2][512]
__global__ void pad_bf16_kernel(const float* __restrict__ x, bf16* __restrict__ xp, int L) {
  long i = (long)blockIdx.x * blockDim.x + threadIdx.x;
  long total = (long)NB * (L + 2) * DM;
  if (i >= total) return;
  int  d  = (int)(i % DM);
  long r  = i / DM;
  int  lp = (int)(r % (L + 2));
  int  b  = (int)(r / (L + 2));
  int  ls = (lp - 1 + L) % L;
  xp[i] = (bf16)x[((long)(b * L + ls)) * DM + d];
}

// wt[d][j*512+c] = (bf16) w[(d*512+c)*3 + j]   (conv weight -> [N=512][K=1536])
__global__ void convw_kernel(const float* __restrict__ w, bf16* __restrict__ wt) {
  int i = blockIdx.x * blockDim.x + threadIdx.x;
  if (i >= DM * 1536) return;
  int d = i / 1536, kk = i % 1536;
  int j = kk / DM, c = kk % DM;
  wt[i] = (bf16)w[(d * DM + c) * 3 + j];
}

__global__ void bnelu_kernel(float* __restrict__ y, const float* __restrict__ g,
                             const float* __restrict__ be, long n) {
  long i = (long)blockIdx.x * blockDim.x + threadIdx.x;
  if (i >= n) return;
  int d = (int)(i % DM);
  float t = y[i] * 0.9999950000374997f * g[d] + be[d];  // 1/sqrt(1+1e-5)
  y[i] = t > 0.f ? t : (__expf(t) - 1.0f);
}

__global__ void maxpool_kernel(const float* __restrict__ y, float* __restrict__ x, int L) {
  int Lo = L >> 1;
  long i = (long)blockIdx.x * blockDim.x + threadIdx.x;
  long total = (long)NB * Lo * DM;
  if (i >= total) return;
  int  d  = (int)(i % DM);
  long r  = i / DM;
  int  lo = (int)(r % Lo);
  int  b  = (int)(r / Lo);
  float m = -3.4e38f;
  for (int t = -1; t <= 1; ++t) {
    int l = 2 * lo + t;
    if (l >= 0 && l < L) m = fmaxf(m, y[((long)(b * L + l)) * DM + d]);
  }
  x[i] = m;
}

__global__ void maxseq_kernel(const float* __restrict__ x, float* __restrict__ p, int L) {
  int i = blockIdx.x * blockDim.x + threadIdx.x;
  if (i >= NB * DM) return;
  int d = i % DM, b = i / DM;
  float m = -3.4e38f;
  for (int l = 0; l < L; ++l) m = fmaxf(m, x[((long)(b * L + l)) * DM + d]);
  p[i] = m;
}

__global__ void proj_kernel(const float* __restrict__ p, const float* __restrict__ w,
                            const float* __restrict__ bb, float* __restrict__ out) {
  int i = threadIdx.x;
  if (i >= NB * 2) return;
  int c = i & 1, b = i >> 1;
  float s = bb[c];
  for (int d = 0; d < DM; ++d) s += p[b * DM + d] * w[c * DM + d];
  out[i] = s;
}

// ---------------------------------------------------------------------------
static inline unsigned gb(long n) { return (unsigned)((n + 255) / 256); }

extern "C" void kernel_launch(void* const* d_in, const int* in_sizes, int n_in,
                              void* d_out, int out_size, void* d_ws, size_t ws_size,
                              hipStream_t stream) {
  (void)in_sizes; (void)n_in; (void)out_size; (void)ws_size;
  const float* xe   = (const float*)d_in[0];
  const float* tokw = (const float*)d_in[1];
  const float* Wq   = (const float*)d_in[2];
  const float* Wk   = (const float*)d_in[3];
  const float* Wv   = (const float*)d_in[4];
  const float* Wo   = (const float*)d_in[5];
  const float* bq   = (const float*)d_in[6];
  const float* bk   = (const float*)d_in[7];
  const float* bv   = (const float*)d_in[8];
  const float* bo   = (const float*)d_in[9];
  const float* W1   = (const float*)d_in[10];
  const float* b1   = (const float*)d_in[11];
  const float* W2   = (const float*)d_in[12];
  const float* b2   = (const float*)d_in[13];
  const float* ln1g = (const float*)d_in[14];
  const float* ln1b = (const float*)d_in[15];
  const float* ln2g = (const float*)d_in[16];
  const float* ln2b = (const float*)d_in[17];
  const float* dcw  = (const float*)d_in[18];
  const float* dcb  = (const float*)d_in[19];
  const float* bng  = (const float*)d_in[20];
  const float* bnb  = (const float*)d_in[21];
  const float* lnfg = (const float*)d_in[22];
  const float* lnfb = (const float*)d_in[23];
  const float* pw   = (const float*)d_in[24];
  const float* pb   = (const float*)d_in[25];

  char* ws = (char*)d_ws;
  size_t off = 0;
  auto alloc = [&](size_t bytes) -> void* {
    void* p = ws + off;
    off += (bytes + 255) & ~(size_t)255;
    return p;
  };
  const size_t ACT = (size_t)NB * 2048 * DM * sizeof(float);   // 32 MB
  float* XF  = (float*)alloc(ACT);
  bf16*  XB  = (bf16*) alloc(ACT / 2);
  float* QF  = (float*)alloc(ACT);
  float* KF  = (float*)alloc(ACT);           // also reused as "Y" temp
  float* VF  = (float*)alloc(ACT);
  float* OF  = (float*)alloc(ACT);
  bf16*  PADB  = (bf16*)alloc((size_t)NB * 2050 * DM * sizeof(bf16));
  bf16*  WSLOT = (bf16*)alloc((size_t)DM * 1536 * sizeof(bf16));
  float* MM  = (float*)alloc((size_t)64 * 2048 * sizeof(float));
  int*   TOP = (int*)  alloc((size_t)64 * 40 * sizeof(int));
  float* VM  = (float*)alloc((size_t)NB * DM * sizeof(float));
  float* PL  = (float*)alloc((size_t)NB * DM * sizeof(float));

  int L = 2048;
  {
    long n = (long)NB * L * DM;
    tok_embed_kernel<<<gb(n), 256, 0, stream>>>(xe, tokw, XF, L);
  }

  for (int i = 0; i < 3; ++i) {
    const int U = (i == 0) ? 40 : 35;
    const long M = (long)NB * L;
    const long n = M * DM;
    const long WN = (long)DM * DM;               // 262144
    dim3 ggrid((unsigned)(M / 128), 4, 1);

    // ---- QKV projections (WMMA GEMMs) ----
    f2bf_kernel<<<gb(n), 256, 0, stream>>>(XF, XB, n);
    f2bf_kernel<<<gb(WN), 256, 0, stream>>>(Wq + (size_t)i * WN, WSLOT, WN);
    gemm_wmma_bf16<<<ggrid, 256, 0, stream>>>(XB, 0, DM, WSLOT, QF, 0, DM, bq + i * DM, DM);
    f2bf_kernel<<<gb(WN), 256, 0, stream>>>(Wk + (size_t)i * WN, WSLOT, WN);
    gemm_wmma_bf16<<<ggrid, 256, 0, stream>>>(XB, 0, DM, WSLOT, KF, 0, DM, bk + i * DM, DM);
    f2bf_kernel<<<gb(WN), 256, 0, stream>>>(Wv + (size_t)i * WN, WSLOT, WN);
    gemm_wmma_bf16<<<ggrid, 256, 0, stream>>>(XB, 0, DM, WSLOT, VF, 0, DM, bv + i * DM, DM);

    // ---- ProbSparse attention ----
    meas_kernel<<<gb((long)64 * L), 256, 0, stream>>>(QF, KF, MM, L, U, i);
    topk_kernel<<<64, 256, 0, stream>>>(MM, TOP, L, U);
    vmean_kernel<<<gb(NB * DM), 256, 0, stream>>>(VF, VM, L);
    ctxfill_kernel<<<gb(n), 256, 0, stream>>>(VM, OF, L);
    attn_update_kernel<<<64 * U, 256, 0, stream>>>(QF, KF, VF, TOP, OF, L, U);

    // ---- output projection + residual LN ----
    f2bf_kernel<<<gb(n), 256, 0, stream>>>(OF, XB, n);
    f2bf_kernel<<<gb(WN), 256, 0, stream>>>(Wo + (size_t)i * WN, WSLOT, WN);
    gemm_wmma_bf16<<<ggrid, 256, 0, stream>>>(XB, 0, DM, WSLOT, KF, 0, DM, bo + i * DM, DM);
    resid_ln_kernel<<<(unsigned)M, 256, 0, stream>>>(XF, KF, XF, ln1g + i * DM, ln1b + i * DM);

    // ---- FFN ----
    f2bf_kernel<<<gb(n), 256, 0, stream>>>(XF, XB, n);
    f2bf_kernel<<<gb(WN), 256, 0, stream>>>(W1 + (size_t)i * WN, WSLOT, WN);
    gemm_wmma_bf16<<<ggrid, 256, 0, stream>>>(XB, 0, DM, WSLOT, KF, 0, DM, b1 + i * DM, DM);
    gelu_kernel<<<gb(n), 256, 0, stream>>>(KF, n);
    f2bf_kernel<<<gb(n), 256, 0, stream>>>(KF, XB, n);
    f2bf_kernel<<<gb(WN), 256, 0, stream>>>(W2 + (size_t)i * WN, WSLOT, WN);
    gemm_wmma_bf16<<<ggrid, 256, 0, stream>>>(XB, 0, DM, WSLOT, QF, 0, DM, b2 + i * DM, DM);
    resid_ln_kernel<<<(unsigned)M, 256, 0, stream>>>(XF, QF, XF, ln2g + i * DM, ln2b + i * DM);

    // ---- distil conv (K=1536 batched WMMA GEMM over circularly padded x) ----
    if (i < 2) {
      long np = (long)NB * (L + 2) * DM;
      pad_bf16_kernel<<<gb(np), 256, 0, stream>>>(XF, PADB, L);
      convw_kernel<<<gb((long)DM * 1536), 256, 0, stream>>>(dcw + (size_t)i * DM * DM * 3, WSLOT);
      dim3 cgrid((unsigned)(L / 128), 4, NB);
      gemm_wmma_bf16<<<cgrid, 256, 0, stream>>>(PADB, (long)(L + 2) * DM, DM,
                                                WSLOT, KF, (long)L * DM, DM,
                                                dcb + i * DM, 1536);
      bnelu_kernel<<<gb(n), 256, 0, stream>>>(KF, bng + i * DM, bnb + i * DM, n);
      long no = (long)NB * (L / 2) * DM;
      maxpool_kernel<<<gb(no), 256, 0, stream>>>(KF, XF, L);
      L >>= 1;
    }
  }

  // ---- final LN, seq-max pool, projection ----
  {
    long M = (long)NB * L;
    resid_ln_kernel<<<(unsigned)M, 256, 0, stream>>>(XF, nullptr, XF, lnfg, lnfb);
    maxseq_kernel<<<gb(NB * DM), 256, 0, stream>>>(XF, PL, L);
    proj_kernel<<<1, 32, 0, stream>>>(PL, pw, pb, (float*)d_out);
  }
}
